// SwinTransformerBlock_46969762349762
// MI455X (gfx1250) — matrix operand order repaired and checked by reading, not compiled, not run on hardware
//
#include <hip/hip_runtime.h>
#include <hip/hip_bf16.h>
#include <math.h>

// ---------------------------------------------------------------- types
typedef __attribute__((ext_vector_type(16))) _Float16 v16h;
typedef __attribute__((ext_vector_type(8)))  _Float16 v8h;
typedef __attribute__((ext_vector_type(8)))  float    v8f;

// ---------------------------------------------------------------- constants
#define Bz    64
#define HH    28
#define WW    28
#define Dm    384
#define HEADS 12
#define WS    7
#define SHIFT 3
#define NTOK  49              // tokens per window
#define NWIN  16              // windows per image (4x4)
#define TOK   50176           // B * H * W == B * nW * N
#define HD    32              // head dim
#define SCALE 0.17677669529663687f   // 32^-0.5

// workspace byte offsets (lifetime-aliased)
#define OFF_XW   ((size_t)0)            // f16 [TOK,384]   LN1+shift+partition
#define OFF_QB   ((size_t)38535168)     // f16 [TOK,384]   q (pre-scaled)
#define OFF_KB   ((size_t)77070336)     // f16 [TOK,384]   k
#define OFF_VT   ((size_t)115605504)    // f16 [1024*12*32, 64]  v transposed, m padded to 64
#define OFF_OB   ((size_t)165937152)    // f16 [TOK,384]   attention out
#define OFF_BM   ((size_t)204472320)    // f32 [16,12,49,49] bias+mask
#define OFF_X1   ((size_t)206316288)    // f32 [TOK,384]   shortcut + attn branch
#define OFF_W1T  ((size_t)283386624)    // f16 [1152,384]  qkv_w^T
#define OFF_W2T  ((size_t)284271360)    // f16 [384,384]   proj_w^T
#define OFF_W3T  ((size_t)284566272)    // f16 [1536,384]  fc1_w^T
#define OFF_W4T  ((size_t)285745920)    // f16 [384,1536]  fc2_w^T
#define OFF_GB   OFF_QB                 // f16 [TOK,1536]  gelu(fc1) (reuses Q/K/VT region)
#define OFF_MB   OFF_XW                 // f16 [TOK,384]   LN2 out (reuses xw region)

static __device__ __forceinline__ int imin(int a, int b) { return a < b ? a : b; }

// ---------------------------------------------------------------- fragment helpers
static __device__ __forceinline__ v16h pack16(v8h lo, v8h hi) {
  v16h f;
#pragma unroll
  for (int i = 0; i < 8; ++i) { f[i] = lo[i]; f[i + 8] = hi[i]; }
  return f;
}
// A 16x32 f16 fragment. rowp = &A[row_of_this_lane][k0].
// lanes 0-15: elems 0..7 = K 0..7, 8..15 = K 16..23; lanes 16-31: +8.
static __device__ __forceinline__ v16h load_a_lane(const _Float16* rowp, int g) {
  const _Float16* p = rowp + g * 8;
  return pack16(*(const v8h*)p, *(const v8h*)(p + 16));
}
// B 32x16 f16 fragment, loaded from B^T row-major. rowp = &Bt[col_of_this_lane][k0].
// lanes 0-15: elems 0..7 = K 0..7, 8..15 = K 16..23; lanes 16-31: +8.
static __device__ __forceinline__ v16h load_b_lane(const _Float16* rowp, int g) {
  const _Float16* p = rowp + g * 8;
  return pack16(*(const v8h*)p, *(const v8h*)(p + 16));
}
static __device__ __forceinline__ v8f wmma16(v16h a, v16h b, v8f c) {
  return __builtin_amdgcn_wmma_f32_16x16x32_f16(false, a, false, b, (short)0, c,
                                                false, false);
}

// ---------------------------------------------------------------- reductions
static __device__ __forceinline__ float sum32(float v) {
#pragma unroll
  for (int m = 16; m >= 1; m >>= 1) v += __shfl_xor(v, m, 32);
  return v;
}
static __device__ __forceinline__ float sum16(float v) {
#pragma unroll
  for (int m = 8; m >= 1; m >>= 1) v += __shfl_xor(v, m, 32);
  return v;
}
static __device__ __forceinline__ float max16(float v) {
#pragma unroll
  for (int m = 8; m >= 1; m >>= 1) v = fmaxf(v, __shfl_xor(v, m, 32));
  return v;
}

// ---------------------------------------------------------------- weight convert+transpose
__global__ void cvt_t_kernel(const float* __restrict__ W, _Float16* __restrict__ Wt,
                             int K, int N) {
  int idx = blockIdx.x * 256 + threadIdx.x;
  if (idx >= K * N) return;
  int k = idx / N, n = idx % N;
  Wt[(size_t)n * K + k] = (_Float16)W[idx];
}

// ---------------------------------------------------------------- bias + shift-mask table
__global__ void prep_bm_kernel(const float* __restrict__ rpb, float* __restrict__ bm) {
  int idx = blockIdx.x * 256 + threadIdx.x;
  if (idx >= NWIN * HEADS * NTOK * NTOK) return;
  int m = idx % NTOK;
  int t = idx / NTOK;
  int n = t % NTOK; t /= NTOK;
  int h = t % HEADS;
  int wi = t / HEADS;
  int nh = n / WS, nw = n % WS, mh = m / WS, mw = m % WS;
  int rel = (nh - mh + WS - 1) * (2 * WS - 1) + (nw - mw + WS - 1);
  float bias = rpb[rel * HEADS + h];
  int wr = wi >> 2, wc = wi & 3;
  int hn = wr * WS + nh, wn = wc * WS + nw;
  int hm = wr * WS + mh, wm = wc * WS + mw;
  int rhn = hn < (HH - WS) ? 0 : (hn < (HH - SHIFT) ? 1 : 2);
  int rwn = wn < (WW - WS) ? 0 : (wn < (WW - SHIFT) ? 1 : 2);
  int rhm = hm < (HH - WS) ? 0 : (hm < (HH - SHIFT) ? 1 : 2);
  int rwm = wm < (WW - WS) ? 0 : (wm < (WW - SHIFT) ? 1 : 2);
  int cn = rhn * 3 + rwn, cm = rhm * 3 + rwm;
  bm[idx] = bias + ((cn != cm) ? -100.0f : 0.0f);
}

// ---------------------------------------------------------------- token -> spatial mapping
static __device__ __forceinline__ int tok_to_spatial(int t) {
  // window-domain token t -> spatial row index of x (undoing shift+partition)
  int w = t / NTOK, n = t % NTOK;
  int bz = w >> 4, wi = w & 15;
  int wr = wi >> 2, wc = wi & 3;
  int nh = n / WS, nw = n % WS;
  int h0 = wr * WS + nh + SHIFT; if (h0 >= HH) h0 -= HH;
  int w0 = wc * WS + nw + SHIFT; if (w0 >= WW) w0 -= WW;
  return bz * (HH * WW) + h0 * WW + w0;
}

// ---------------------------------------------------------------- LN1 + shift + partition
__global__ __launch_bounds__(256) void ln1_kernel(const float* __restrict__ x,
                                                  const float* __restrict__ gam,
                                                  const float* __restrict__ bet,
                                                  _Float16* __restrict__ xw) {
  int wave = threadIdx.x >> 5, lane = threadIdx.x & 31;
  int t = blockIdx.x * 8 + wave;
  const float* xr = x + (size_t)tok_to_spatial(t) * Dm;
  float v[12], s = 0.f, sq = 0.f;
#pragma unroll
  for (int j = 0; j < 12; ++j) {
    v[j] = xr[lane + 32 * j];
    s += v[j]; sq += v[j] * v[j];
  }
  s = sum32(s); sq = sum32(sq);
  float mean = s * (1.0f / Dm);
  float var = sq * (1.0f / Dm) - mean * mean;
  float rinv = rsqrtf(var + 1e-5f);
#pragma unroll
  for (int j = 0; j < 12; ++j) {
    int c = lane + 32 * j;
    xw[(size_t)t * Dm + c] = (_Float16)((v[j] - mean) * rinv * gam[c] + bet[c]);
  }
}

// ---------------------------------------------------------------- LN2 (identity layout)
__global__ __launch_bounds__(256) void ln2_kernel(const float* __restrict__ x1,
                                                  const float* __restrict__ gam,
                                                  const float* __restrict__ bet,
                                                  _Float16* __restrict__ mb) {
  int wave = threadIdx.x >> 5, lane = threadIdx.x & 31;
  int t = blockIdx.x * 8 + wave;
  const float* xr = x1 + (size_t)t * Dm;
  float v[12], s = 0.f, sq = 0.f;
#pragma unroll
  for (int j = 0; j < 12; ++j) {
    v[j] = xr[lane + 32 * j];
    s += v[j]; sq += v[j] * v[j];
  }
  s = sum32(s); sq = sum32(sq);
  float mean = s * (1.0f / Dm);
  float var = sq * (1.0f / Dm) - mean * mean;
  float rinv = rsqrtf(var + 1e-5f);
#pragma unroll
  for (int j = 0; j < 12; ++j) {
    int c = lane + 32 * j;
    mb[(size_t)t * Dm + c] = (_Float16)((v[j] - mean) * rinv * gam[c] + bet[c]);
  }
}

// ---------------------------------------------------------------- GEMM epilogues
struct EpiQKV {
  _Float16* Qb; _Float16* Kb; _Float16* VT; const float* bias;
  __device__ void operator()(int row, int col, float v) const {
    v += bias[col];
    int i = col / Dm, wi = col % Dm;
    if (i == 0) {
      Qb[(size_t)row * Dm + wi] = (_Float16)(v * SCALE);
    } else if (i == 1) {
      Kb[(size_t)row * Dm + wi] = (_Float16)v;
    } else {
      int h = wi >> 5, d = wi & 31;
      int w = row / NTOK, n = row % NTOK;
      VT[(((size_t)w * HEADS + h) * HD + d) * 64 + n] = (_Float16)v;
    }
  }
};
struct EpiProj {
  const float* bias; const float* x; float* x1;
  __device__ void operator()(int row, int col, float v) const {
    v += bias[col];
    size_t o = (size_t)tok_to_spatial(row) * Dm + col;
    x1[o] = x[o] + v;
  }
};
struct EpiFC1 {
  const float* bias; _Float16* g;
  __device__ void operator()(int row, int col, float v) const {
    v += bias[col];
    float gl = 0.5f * v * (1.0f + erff(v * 0.70710678118654752f));
    g[(size_t)row * 1536 + col] = (_Float16)gl;
  }
};
struct EpiFC2 {
  const float* bias; const float* x1; float* out;
  __device__ void operator()(int row, int col, float v) const {
    v += bias[col];
    size_t o = (size_t)row * Dm + col;
    out[o] = x1[o] + v;
  }
};

// ---------------------------------------------------------------- generic WMMA GEMM
// C[M,N] = A[M,K](f16,row-major) * Bt[N,K]^T + epilogue.
// block = 256 thr (8 waves as 2x4); block tile 128x128; wave tile 64x32.
template <class Epi>
__global__ __launch_bounds__(256) void gemm_f16(const _Float16* __restrict__ A,
                                                const _Float16* __restrict__ Bt,
                                                int K, Epi epi) {
  int lane = threadIdx.x & 31, wave = threadIdx.x >> 5;
  int r15 = lane & 15, g = lane >> 4;
  int m0 = blockIdx.y * 128 + (wave >> 2) * 64;
  int n0 = blockIdx.x * 128 + (wave & 3) * 32;
  const _Float16* Arow[4];
  const _Float16* Brow[2];
#pragma unroll
  for (int mt = 0; mt < 4; ++mt) Arow[mt] = A + (size_t)(m0 + mt * 16 + r15) * K;
#pragma unroll
  for (int nt = 0; nt < 2; ++nt) Brow[nt] = Bt + (size_t)(n0 + nt * 16 + r15) * K;
  v8f acc[4][2] = {};
  for (int k0 = 0; k0 < K; k0 += 32) {
    v16h a[4], b[2];
#pragma unroll
    for (int mt = 0; mt < 4; ++mt) a[mt] = load_a_lane(Arow[mt] + k0, g);
#pragma unroll
    for (int nt = 0; nt < 2; ++nt) b[nt] = load_b_lane(Brow[nt] + k0, g);
#pragma unroll
    for (int mt = 0; mt < 4; ++mt)
#pragma unroll
      for (int nt = 0; nt < 2; ++nt)
        acc[mt][nt] = wmma16(a[mt], b[nt], acc[mt][nt]);
  }
#pragma unroll
  for (int mt = 0; mt < 4; ++mt)
#pragma unroll
    for (int nt = 0; nt < 2; ++nt)
#pragma unroll
      for (int e = 0; e < 8; ++e)
        epi(m0 + mt * 16 + e + 8 * g, n0 + nt * 16 + r15, acc[mt][nt][e]);
}

// ---------------------------------------------------------------- attention
// one block per (window, head); 4 waves, wave owns a 16-row strip of the 49(pad64)
// score matrix. QK^T (4 wmma) -> bias/mask -> softmax (16-lane shuffles) ->
// P via LDS -> P*V (4 wmma) -> store O.
__global__ __launch_bounds__(128) void attn_kernel(const _Float16* __restrict__ Qb,
                                                   const _Float16* __restrict__ Kb,
                                                   const _Float16* __restrict__ VT,
                                                   const float* __restrict__ bm,
                                                   _Float16* __restrict__ Ob) {
  __shared__ _Float16 Pl[4 * 16 * 64];
  int blk = blockIdx.x;
  int w = blk / HEADS, h = blk % HEADS;
  int wi = w & 15;
  int wave = threadIdx.x >> 5, lane = threadIdx.x & 31;
  int r15 = lane & 15, g = lane >> 4;

  // ---- S = (q*SCALE) k^T
  int nrow = wave * 16 + r15;
  const _Float16* qrow = Qb + ((size_t)w * NTOK + imin(nrow, NTOK - 1)) * Dm + h * HD;
  v16h a = load_a_lane(qrow, g);
  v8f s[4];
  const v8f zero = {};
#pragma unroll
  for (int nt = 0; nt < 4; ++nt) {
    int m = nt * 16 + r15;
    const _Float16* krow = Kb + ((size_t)w * NTOK + imin(m, NTOK - 1)) * Dm + h * HD;
    v16h b = load_b_lane(krow, g);
    s[nt] = wmma16(a, b, zero);
  }

  // ---- + relpos bias + shift mask, column padding -> -inf
  const float* bmbase = bm + (size_t)(wi * HEADS + h) * NTOK * NTOK;
  float sc[4][8];
#pragma unroll
  for (int nt = 0; nt < 4; ++nt)
#pragma unroll
    for (int e = 0; e < 8; ++e) {
      int row = wave * 16 + e + 8 * g;
      int col = nt * 16 + r15;
      float val = s[nt][e] +
                  bmbase[imin(row, NTOK - 1) * NTOK + imin(col, NTOK - 1)];
      if (col >= NTOK) val = -1e30f;
      sc[nt][e] = val;
    }

  // ---- softmax per row (each row's 64 cols live across 16 lanes x 4 accs)
#pragma unroll
  for (int e = 0; e < 8; ++e) {
    float mx = fmaxf(fmaxf(sc[0][e], sc[1][e]), fmaxf(sc[2][e], sc[3][e]));
    mx = max16(mx);
    float ssum = 0.f;
#pragma unroll
    for (int nt = 0; nt < 4; ++nt) {
      float p = __expf(sc[nt][e] - mx);
      sc[nt][e] = p;
      ssum += p;
    }
    ssum = sum16(ssum);
    float inv = 1.0f / ssum;
#pragma unroll
    for (int nt = 0; nt < 4; ++nt) sc[nt][e] *= inv;
  }

  // ---- P -> LDS (per-wave private 16x64 region) then O = P * V
  _Float16* Pw = Pl + wave * (16 * 64);
#pragma unroll
  for (int nt = 0; nt < 4; ++nt)
#pragma unroll
    for (int e = 0; e < 8; ++e)
      Pw[(e + 8 * g) * 64 + nt * 16 + r15] = (_Float16)sc[nt][e];

  v8f o[2] = {};
  const _Float16* vth = VT + ((size_t)w * HEADS + h) * HD * 64;
#pragma unroll
  for (int kt = 0; kt < 2; ++kt) {
    v16h pa = load_a_lane(Pw + r15 * 64 + kt * 32, g);
#pragma unroll
    for (int nt2 = 0; nt2 < 2; ++nt2) {
      v16h vb = load_b_lane(vth + (size_t)(nt2 * 16 + r15) * 64 + kt * 32, g);
      o[nt2] = wmma16(pa, vb, o[nt2]);
    }
  }

  // ---- store O rows < 49
#pragma unroll
  for (int nt2 = 0; nt2 < 2; ++nt2)
#pragma unroll
    for (int e = 0; e < 8; ++e) {
      int n = wave * 16 + e + 8 * g;
      if (n < NTOK) {
        int d = nt2 * 16 + r15;
        Ob[((size_t)w * NTOK + n) * Dm + h * HD + d] = (_Float16)o[nt2][e];
      }
    }
}

// ---------------------------------------------------------------- launch
extern "C" void kernel_launch(void* const* d_in, const int* in_sizes, int n_in,
                              void* d_out, int out_size, void* d_ws, size_t ws_size,
                              hipStream_t stream) {
  const float* x       = (const float*)d_in[0];
  const float* n1g     = (const float*)d_in[1];
  const float* n1b     = (const float*)d_in[2];
  const float* qkv_w   = (const float*)d_in[3];
  const float* qkv_b   = (const float*)d_in[4];
  const float* proj_w  = (const float*)d_in[5];
  const float* proj_b  = (const float*)d_in[6];
  const float* rpb     = (const float*)d_in[7];
  const float* n2g     = (const float*)d_in[8];
  const float* n2b     = (const float*)d_in[9];
  const float* fc1_w   = (const float*)d_in[10];
  const float* fc1_b   = (const float*)d_in[11];
  const float* fc2_w   = (const float*)d_in[12];
  const float* fc2_b   = (const float*)d_in[13];
  float* out = (float*)d_out;

  char* ws = (char*)d_ws;
  _Float16* xw   = (_Float16*)(ws + OFF_XW);
  _Float16* Qb   = (_Float16*)(ws + OFF_QB);
  _Float16* Kb   = (_Float16*)(ws + OFF_KB);
  _Float16* VT   = (_Float16*)(ws + OFF_VT);
  _Float16* Ob   = (_Float16*)(ws + OFF_OB);
  float*    bmT  = (float*)   (ws + OFF_BM);
  float*    x1   = (float*)   (ws + OFF_X1);
  _Float16* w1t  = (_Float16*)(ws + OFF_W1T);
  _Float16* w2t  = (_Float16*)(ws + OFF_W2T);
  _Float16* w3t  = (_Float16*)(ws + OFF_W3T);
  _Float16* w4t  = (_Float16*)(ws + OFF_W4T);
  _Float16* gb   = (_Float16*)(ws + OFF_GB);
  _Float16* mb   = (_Float16*)(ws + OFF_MB);

  // 1. convert + transpose weights to f16 [N,K]
  cvt_t_kernel<<<(Dm * 1152 + 255) / 256, 256, 0, stream>>>(qkv_w, w1t, Dm, 1152);
  cvt_t_kernel<<<(Dm * Dm   + 255) / 256, 256, 0, stream>>>(proj_w, w2t, Dm, Dm);
  cvt_t_kernel<<<(Dm * 1536 + 255) / 256, 256, 0, stream>>>(fc1_w, w3t, Dm, 1536);
  cvt_t_kernel<<<(1536 * Dm + 255) / 256, 256, 0, stream>>>(fc2_w, w4t, 1536, Dm);

  // 2. fused bias+mask table
  prep_bm_kernel<<<(NWIN * HEADS * NTOK * NTOK + 255) / 256, 256, 0, stream>>>(rpb, bmT);

  // 3. LN1 + cyclic shift + window partition
  ln1_kernel<<<TOK / 8, 256, 0, stream>>>(x, n1g, n1b, xw);

  // 4. QKV gemm: [50176,384] x [384,1152]
  {
    EpiQKV e{Qb, Kb, VT, qkv_b};
    gemm_f16<EpiQKV><<<dim3(1152 / 128, TOK / 128), 256, 0, stream>>>(xw, w1t, Dm, e);
  }

  // 5. windowed attention: 1024 windows x 12 heads
  attn_kernel<<<1024 * HEADS, 128, 0, stream>>>(Qb, Kb, VT, bmT, Ob);

  // 6. proj gemm + window reverse + unshift + residual -> x1
  {
    EpiProj e{proj_b, x, x1};
    gemm_f16<EpiProj><<<dim3(Dm / 128, TOK / 128), 256, 0, stream>>>(Ob, w2t, Dm, e);
  }

  // 7. LN2
  ln2_kernel<<<TOK / 8, 256, 0, stream>>>(x1, n2g, n2b, mb);

  // 8. FC1 + exact GELU
  {
    EpiFC1 e{fc1_b, gb};
    gemm_f16<EpiFC1><<<dim3(1536 / 128, TOK / 128), 256, 0, stream>>>(mb, w3t, Dm, e);
  }

  // 9. FC2 + residual -> out
  {
    EpiFC2 e{fc2_b, x1, out};
    gemm_f16<EpiFC2><<<dim3(Dm / 128, TOK / 128), 256, 0, stream>>>(gb, w4t, 1536, e);
  }
}
